// GraphConvLayer_18734647345262
// MI455X (gfx1250) — matrix operand order repaired and checked by reading, not compiled, not run on hardware
//
#include <hip/hip_runtime.h>
#include <hip/hip_bf16.h>
#include <stdint.h>

typedef __attribute__((ext_vector_type(16))) __bf16 v16bf;
typedef __attribute__((ext_vector_type(8)))  __bf16 v8bf;
typedef __attribute__((ext_vector_type(8)))  float  v8f;

#define LN_EPS 1e-5f

// float -> bf16 with round-to-nearest-even (bit trick)
static __device__ inline unsigned short f2bf(float x) {
  uint32_t u = __builtin_bit_cast(uint32_t, x);
  u += 0x7FFFu + ((u >> 16) & 1u);
  return (unsigned short)(u >> 16);
}

// ---------------------------------------------------------------------------
// Kernel 1: LayerNorm over d=256 + cast to bf16. One wave32 per row.
// Rows = b*n = 2048. 256 threads/block = 8 waves => 8 rows/block, 256 blocks.
// ---------------------------------------------------------------------------
__global__ void ln_cast_kernel(const float* __restrict__ H,
                               const float* __restrict__ gamma,
                               const float* __restrict__ beta,
                               unsigned short* __restrict__ Hn) {
  const int d = 256;
  int wave = threadIdx.x >> 5;
  int lane = threadIdx.x & 31;
  int row  = blockIdx.x * 8 + wave;            // 0..2047, exact cover
  const float* h = H + (size_t)row * d;

  float x[8];
  float s = 0.f;
#pragma unroll
  for (int j = 0; j < 8; ++j) { x[j] = h[lane + 32 * j]; s += x[j]; }
#pragma unroll
  for (int m = 16; m >= 1; m >>= 1) s += __shfl_xor(s, m, 32);
  float mu = s * (1.0f / 256.0f);

  float q = 0.f;
#pragma unroll
  for (int j = 0; j < 8; ++j) { float t = x[j] - mu; q += t * t; }
#pragma unroll
  for (int m = 16; m >= 1; m >>= 1) q += __shfl_xor(q, m, 32);
  float rstd = rsqrtf(q * (1.0f / 256.0f) + LN_EPS);

  unsigned short* o = Hn + (size_t)row * d;
#pragma unroll
  for (int j = 0; j < 8; ++j) {
    int c = lane + 32 * j;
    float y = (x[j] - mu) * rstd * gamma[c] + beta[c];
    o[c] = f2bf(y);
  }
}

// ---------------------------------------------------------------------------
// Kernel 2: cast W [256x256] to bf16.  65536 elems -> 256 blocks x 256 thr.
// ---------------------------------------------------------------------------
__global__ void cast_w_kernel(const float* __restrict__ W,
                              unsigned short* __restrict__ Wb) {
  int i = blockIdx.x * 256 + threadIdx.x;
  Wb[i] = f2bf(W[i]);
}

// ---------------------------------------------------------------------------
// Kernel 3: weights = coeff * A, cast to bf16. 262144 elems -> 1024 blocks.
// ---------------------------------------------------------------------------
__global__ void cast_adj_kernel(const float* __restrict__ coeff,
                                const float* __restrict__ A,
                                unsigned short* __restrict__ Adjb) {
  int i = blockIdx.x * 256 + threadIdx.x;
  Adjb[i] = f2bf(coeff[i] * A[i]);
}

// ---------------------------------------------------------------------------
// Wave-level 16(M) x 64(N) tile MACC over K using v_wmma_f32_16x16x32_bf16.
//   A  : [M x K] bf16 row-major (row = output row)
//   Bt : [N x K] bf16 row-major (row = output col; i.e. B-transposed)
// A-frag (16x32): lane m=lane&15, K group = (lane>>4)*16 -> two 16B chunks
// (K 16h+0..7 and 16h+8..15), matching B's contiguous order.
// B-frag (32x16): lane n=lane&15, K group = (lane>>4)*16 -> one 32B row chunk.
// ---------------------------------------------------------------------------
__device__ inline void wave_tile_16x64(const unsigned short* __restrict__ A,
                                       const unsigned short* __restrict__ Bt,
                                       int K, int m0, int n0, v8f acc[4]) {
  int lane  = threadIdx.x & 31;
  int lrow  = lane & 15;
  int lhalf = lane >> 4;
  const unsigned short* arow = A  + (size_t)(m0 + lrow) * K + lhalf * 16;
  const unsigned short* brow = Bt + (size_t)(n0 + lrow) * K + lhalf * 16;

  for (int k0 = 0; k0 < K; k0 += 32) {
    __builtin_prefetch(arow + k0 + 64, 0, 0);   // -> global_prefetch_b8
    v8bf alo = *(const v8bf*)(arow + k0);
    v8bf ahi = *(const v8bf*)(arow + k0 + 8);
    v16bf af;
#pragma unroll
    for (int i = 0; i < 8; ++i) { af[i] = alo[i]; af[i + 8] = ahi[i]; }
#pragma unroll
    for (int t = 0; t < 4; ++t) {
      v16bf bf = *(const v16bf*)(brow + (size_t)t * 16 * K + k0);
      acc[t] = __builtin_amdgcn_wmma_f32_16x16x32_bf16(
          /*neg_a=*/false, af, /*neg_b=*/false, bf,
          /*c_mod=*/(short)0, acc[t], /*reuse_a=*/false, /*reuse_b=*/false);
    }
  }
}

// ---------------------------------------------------------------------------
// Kernel 4: msg = relu(Hn @ W^T + bias), stored TRANSPOSED as
// msgT[b][d=256][m=512] (bf16) so GEMM2's B-operand rows are contiguous.
// M=2048, N=256, K=256. 512 wave-tiles of 16x64 -> 64 blocks x 8 waves.
// ---------------------------------------------------------------------------
__global__ void gemm1_kernel(const unsigned short* __restrict__ Hn,   // [2048,256]
                             const unsigned short* __restrict__ Wb,   // [256,256] = Bt
                             const float* __restrict__ bias,          // [256]
                             unsigned short* __restrict__ msgT) {     // [4][256][512]
  const int K = 256;
  int wave = threadIdx.x >> 5;
  int lane = threadIdx.x & 31;
  int tile = blockIdx.x * 8 + wave;   // 0..511
  int m0 = (tile >> 2) * 16;          // 128 M-tiles
  int n0 = (tile & 3) * 64;           // 4 N-strips

  v8f acc[4] = {};
  wave_tile_16x64(Hn, Wb, K, m0, n0, acc);

  int lrow = lane & 15, lhalf = lane >> 4;
#pragma unroll
  for (int t = 0; t < 4; ++t) {
    int n = n0 + t * 16 + lrow;
    float bn = bias[n];
#pragma unroll
    for (int r = 0; r < 8; ++r) {
      int m  = m0 + lhalf * 8 + r;    // flat row in [0,2048)
      int b  = m >> 9;
      int ml = m & 511;
      float v = acc[t][r] + bn;
      v = v > 0.f ? v : 0.f;
      msgT[((size_t)b * 256 + n) * 512 + ml] = f2bf(v);
    }
  }
}

// ---------------------------------------------------------------------------
// Kernel 5: out[b] = (weights @ msg[b]) / 512, fp32 output.
// Per batch: M=512 (node), N=256 (feat), K=512. A = weights bf16 [512,512],
// Bt = msgT[b] [256,512]. 4*32*4 = 512 wave-tiles -> 64 blocks x 8 waves.
// ---------------------------------------------------------------------------
__global__ void gemm2_kernel(const unsigned short* __restrict__ Adjb,  // [512,512]
                             const unsigned short* __restrict__ msgT,  // [4][256][512]
                             float* __restrict__ out) {                // [4,512,256]
  const int K = 512;
  int wave = threadIdx.x >> 5;
  int lane = threadIdx.x & 31;
  int tile = blockIdx.x * 8 + wave;   // 0..511
  int b  = tile >> 7;                 // 0..3
  int lt = tile & 127;
  int m0 = (lt >> 2) * 16;            // 32 M-tiles
  int n0 = (lt & 3) * 64;             // 4 N-strips
  const unsigned short* Bt = msgT + (size_t)b * 256 * 512;

  v8f acc[4] = {};
  wave_tile_16x64(Adjb, Bt, K, m0, n0, acc);

  const float scale = 1.0f / 512.0f;
  int lrow = lane & 15, lhalf = lane >> 4;
#pragma unroll
  for (int t = 0; t < 4; ++t) {
    int n = n0 + t * 16 + lrow;
#pragma unroll
    for (int r = 0; r < 8; ++r) {
      int m = m0 + lhalf * 8 + r;
      out[((size_t)b * 512 + m) * 256 + n] = acc[t][r] * scale;
    }
  }
}

// ---------------------------------------------------------------------------
// Launch: inputs in setup_inputs() order:
//   0:H [4,512,256] 1:A [512,512] 2:coefficients [512,512]
//   3:ln_gamma [256] 4:ln_beta [256] 5:W [256,256] 6:bias [256]
// Workspace (bf16 elems): Hn 524288 | Wb 65536 | Adjb 262144 | msgT 524288
//   => ~2.75 MB total.
// ---------------------------------------------------------------------------
extern "C" void kernel_launch(void* const* d_in, const int* in_sizes, int n_in,
                              void* d_out, int out_size, void* d_ws, size_t ws_size,
                              hipStream_t stream) {
  const float* H     = (const float*)d_in[0];
  const float* A     = (const float*)d_in[1];
  const float* coeff = (const float*)d_in[2];
  const float* gamma = (const float*)d_in[3];
  const float* beta  = (const float*)d_in[4];
  const float* W     = (const float*)d_in[5];
  const float* bias  = (const float*)d_in[6];
  float* out = (float*)d_out;

  unsigned short* ws   = (unsigned short*)d_ws;
  unsigned short* Hn   = ws;              // 524288 elems
  unsigned short* Wb   = ws + 524288;     //  65536 elems
  unsigned short* Adjb = ws + 589824;     // 262144 elems
  unsigned short* msgT = ws + 851968;     // 524288 elems

  ln_cast_kernel<<<256, 256, 0, stream>>>(H, gamma, beta, Hn);
  cast_w_kernel<<<256, 256, 0, stream>>>(W, Wb);
  cast_adj_kernel<<<1024, 256, 0, stream>>>(coeff, A, Adjb);
  gemm1_kernel<<<64, 256, 0, stream>>>(Hn, Wb, bias, msgT);
  gemm2_kernel<<<64, 256, 0, stream>>>(Adjb, msgT, out);
}
